// InverseNetwork_49452253446730
// MI455X (gfx1250) — matrix operand order repaired and checked by reading, not compiled, not run on hardware
//
#include <hip/hip_runtime.h>

// ---------------- problem constants ----------------
#define BQ   32
#define SQ   1024
#define DQ   512
#define HQ   128
#define KNN  10
#define BS   (BQ * SQ)          // 32768
#define BIGF 1e30f

// ---------------- WMMA types ----------------
typedef __attribute__((ext_vector_type(16))) __bf16 v16bf;
typedef __attribute__((ext_vector_type(8)))  float  v8f;

union AB16 { v16bf v; uint4 u[2]; };

// ---------------- TDM descriptor types ----------------
typedef unsigned int u32x4 __attribute__((ext_vector_type(4)));
typedef int          i32x8 __attribute__((ext_vector_type(8)));
typedef int          i32x4 __attribute__((ext_vector_type(4)));

#if defined(__has_builtin)
#if __has_builtin(__builtin_amdgcn_tensor_load_to_lds)
#define HAVE_TDM 1
#endif
#endif

__device__ __forceinline__ unsigned short f2bf(float f) {
  // round-to-nearest-even f32 -> bf16 (bit level)
  unsigned int u = __float_as_uint(f);
  unsigned int r = (u + 0x7FFFu + ((u >> 16) & 1u)) >> 16;
  return (unsigned short)r;
}

// ---------------- f32 -> bf16 convert ----------------
__global__ void cvt_bf16_kernel(const float* __restrict__ src,
                                unsigned short* __restrict__ dst, int n) {
  int i = blockIdx.x * blockDim.x + threadIdx.x;
  int stride = gridDim.x * blockDim.x;
  for (; i < n; i += stride) dst[i] = f2bf(src[i]);
}

// ---------------- weight transpose + convert: W[K,N] f32 -> Wt[N,K] bf16 ----------------
__global__ __launch_bounds__(256) void transpose_bf16_kernel(
    const float* __restrict__ W, unsigned short* __restrict__ Wt, int K, int N) {
  __shared__ float tile[32][33];
  const int kb = blockIdx.y * 32;
  const int nb = blockIdx.x * 32;
  const int tx = threadIdx.x & 31;
  const int ty = threadIdx.x >> 5;   // 0..7
  #pragma unroll
  for (int r = ty; r < 32; r += 8)
    tile[r][tx] = W[(size_t)(kb + r) * N + nb + tx];
  __syncthreads();
  #pragma unroll
  for (int r = ty; r < 32; r += 8)
    Wt[(size_t)(nb + r) * K + kb + tx] = f2bf(tile[tx][r]);
}

// ---------------- WMMA GEMM + bias + ReLU ----------------
// C[M,N] = relu(A[M,K] * Wt[N,K]^T + bias), bf16 inputs, f32 accumulate.
// Block: 256 threads (8 waves), 128x128 tile, K-step 32, TDM double-buffered.
#define MT 128
#define NT 128
#define KT 32
#define LDA 40   // padded row stride (bf16 elems): 64B data + 16B pad
#define LDB 40

#ifdef HAVE_TDM
// Issue one TDM 2D tile load: 128 rows x 32 cols of bf16, row stride
// `stride_elems`, destination LDS with 4-DWORD pad every 16 DWORDs (64B->80B rows).
__device__ __forceinline__ void tdm_load_tile(unsigned lds_off,
                                              const unsigned short* gptr,
                                              unsigned stride_elems) {
  unsigned long long ga = (unsigned long long)(size_t)gptr;
  u32x4 g0;
  g0[0] = 1u;                                    // count=1 (valid descriptor)
  g0[1] = lds_off;                               // lds_addr (bytes)
  g0[2] = (unsigned)(ga & 0xFFFFFFFFu);          // global_addr[31:0]
  g0[3] = (unsigned)((ga >> 32) & 0x01FFFFFFu)   // global_addr[56:32]
          | (2u << 30);                          // type=2 ("image")
  i32x8 g1;
  g1[0] = (int)((1u << 16)                       // data_size = 2 bytes
                | (1u << 20)                     // pad_enable
                | (3u << 22)                     // pad_interval: 16 DWORDs (64B)
                | (3u << 25));                   // pad_amount: 4 DWORDs (16B)
  g1[1] = (int)(((1u << 20) & 0xFFFFu) << 16);   // tensor_dim0 = 1<<20 (lo16=0 in [31:16])
  g1[2] = (int)((((1u << 20) >> 16) & 0xFFFFu)   // tensor_dim0 hi16 = 0x10
                | (((1u << 20) & 0xFFFFu) << 16)); // tensor_dim1 lo16 = 0
  g1[3] = (int)((((1u << 20) >> 16) & 0xFFFFu)   // tensor_dim1 hi16 = 0x10
                | (32u << 16));                  // tile_dim0 = 32
  g1[4] = (int)(128u);                           // tile_dim1 = 128, tile_dim2 = 0
  g1[5] = (int)stride_elems;                     // tensor_dim0_stride lo32
  g1[6] = 0;                                     // stride hi16 | dim1_stride lo16
  g1[7] = 0;
  i32x4 gz4;
  gz4[0] = 0; gz4[1] = 0; gz4[2] = 0; gz4[3] = 0;
  i32x8 gz8;
  gz8[0] = 0; gz8[1] = 0; gz8[2] = 0; gz8[3] = 0;
  gz8[4] = 0; gz8[5] = 0; gz8[6] = 0; gz8[7] = 0;
  // 6-arg form (amdgpu-toolchain / clang-23 lane)
  __builtin_amdgcn_tensor_load_to_lds(g0, g1, gz4, gz4, gz8, 0);
}
#endif

template <bool OUT_F32>
__global__ __launch_bounds__(256) void gemm_bias_relu(
    const unsigned short* __restrict__ A,    // [M,K] bf16
    const unsigned short* __restrict__ Wt,   // [N,K] bf16 (pre-transposed)
    const float* __restrict__ bias,          // [N]
    unsigned short* __restrict__ out_bf,     // [M,N] if !OUT_F32
    float* __restrict__ out_f,               // [M,N] if OUT_F32
    int M, int N, int Kdim) {
  __shared__ unsigned short As[2][MT * LDA];
  __shared__ unsigned short Bs[2][NT * LDB];

  const int tid  = threadIdx.x;
  const int wave = tid >> 5;
  const int lane = tid & 31;
  const int m0 = blockIdx.y * MT;
  const int n0 = blockIdx.x * NT;
  const int wr = wave >> 2;   // 0..1  -> 64-row strip
  const int wc = wave & 3;    // 0..3  -> 32-col strip
  const int nsteps = Kdim / KT;

  v8f acc[4][2];
  #pragma unroll
  for (int mt = 0; mt < 4; ++mt)
    #pragma unroll
    for (int nt = 0; nt < 2; ++nt)
      #pragma unroll
      for (int c = 0; c < 8; ++c) acc[mt][nt][c] = 0.f;

#ifdef HAVE_TDM
  if (wave == 0) {
    tdm_load_tile((unsigned)(size_t)&As[0][0], A  + (size_t)m0 * Kdim, (unsigned)Kdim);
    tdm_load_tile((unsigned)(size_t)&Bs[0][0], Wt + (size_t)n0 * Kdim, (unsigned)Kdim);
  }
#endif

  for (int s = 0; s < nsteps; ++s) {
    const int cur = s & 1;
#ifdef HAVE_TDM
    if (s + 1 < nsteps) {
      if (wave == 0) {
        tdm_load_tile((unsigned)(size_t)&As[cur ^ 1][0],
                      A + (size_t)m0 * Kdim + (s + 1) * KT, (unsigned)Kdim);
        tdm_load_tile((unsigned)(size_t)&Bs[cur ^ 1][0],
                      Wt + (size_t)n0 * Kdim + (s + 1) * KT, (unsigned)Kdim);
      }
      __builtin_amdgcn_s_wait_tensorcnt(2);  // current tile's pair complete
    } else {
      __builtin_amdgcn_s_wait_tensorcnt(0);
    }
#else
    // fallback: per-thread coalesced copy of both tiles (row-major, no scatter)
    {
      const int kt = s * KT;
      #pragma unroll
      for (int p = 0; p < 2; ++p) {
        int li  = p * 256 + tid;
        int row = li >> 2;
        int kc  = (li & 3) << 3;
        uint4 va = *reinterpret_cast<const uint4*>(A + (size_t)(m0 + row) * Kdim + kt + kc);
        *reinterpret_cast<uint4*>(&As[cur][row * LDA + kc]) = va;
        uint4 vb = *reinterpret_cast<const uint4*>(Wt + (size_t)(n0 + row) * Kdim + kt + kc);
        *reinterpret_cast<uint4*>(&Bs[cur][row * LDB + kc]) = vb;
      }
    }
#endif
    __syncthreads();

    // --- fragments (16-bit A/B ISA layouts) ---
    const int l16 = lane & 15;
    const int ak  = (lane & 16) ? 8 : 0;    // lo lanes K{0-7,16-23}; hi K{8-15,24-31}
    const int bk  = (lane & 16) ? 8 : 0;    // lo lanes K{0-7,16-23}; hi K{8-15,24-31}
    AB16 afr[4], bfr[2];
    #pragma unroll
    for (int mt = 0; mt < 4; ++mt) {
      int r = wr * 64 + mt * 16 + l16;
      afr[mt].u[0] = *reinterpret_cast<const uint4*>(&As[cur][r * LDA + ak]);
      afr[mt].u[1] = *reinterpret_cast<const uint4*>(&As[cur][r * LDA + ak + 16]);
    }
    #pragma unroll
    for (int nt = 0; nt < 2; ++nt) {
      int c = wc * 32 + nt * 16 + l16;
      bfr[nt].u[0] = *reinterpret_cast<const uint4*>(&Bs[cur][c * LDB + bk]);
      bfr[nt].u[1] = *reinterpret_cast<const uint4*>(&Bs[cur][c * LDB + bk + 16]);
    }
    #pragma unroll
    for (int mt = 0; mt < 4; ++mt)
      #pragma unroll
      for (int nt = 0; nt < 2; ++nt)
        acc[mt][nt] = __builtin_amdgcn_wmma_f32_16x16x32_bf16(
            false, afr[mt].v, false, bfr[nt].v, (short)0, acc[mt][nt],
            false, false);
    __syncthreads();
  }

  // --- epilogue: bias + ReLU, 32-bit C/D layout ---
  const int l16 = lane & 15;
  const int mo  = (lane & 16) ? 8 : 0;
  #pragma unroll
  for (int mt = 0; mt < 4; ++mt) {
    #pragma unroll
    for (int nt = 0; nt < 2; ++nt) {
      int n = n0 + wc * 32 + nt * 16 + l16;
      float bv = bias[n];
      #pragma unroll
      for (int v = 0; v < 8; ++v) {
        int m = m0 + wr * 64 + mt * 16 + mo + v;
        float val = fmaxf(acc[mt][nt][v] + bv, 0.f);
        if (OUT_F32) out_f[(size_t)m * N + n] = val;
        else         out_bf[(size_t)m * N + n] = f2bf(val);
      }
    }
  }
}

// ---------------- row squared norms ----------------
__global__ void sq_kernel(const float* __restrict__ E, float* __restrict__ sqv) {
  int bs = blockIdx.x * blockDim.x + threadIdx.x;
  if (bs >= BS) return;
  const float4* e = reinterpret_cast<const float4*>(E + (size_t)bs * HQ);
  float s = 0.f;
  #pragma unroll
  for (int q = 0; q < HQ / 4; ++q) {
    float4 v = e[q];
    s += v.x * v.x + v.y * v.y + v.z * v.z + v.w * v.w;
  }
  sqv[bs] = s;
}

// ---------------- causal pairwise distance + top-K ----------------
__global__ __launch_bounds__(128) void dist_topk_kernel(
    const float* __restrict__ E, const float* __restrict__ sqv,
    float* __restrict__ dists, float* __restrict__ bm) {
  __shared__ float Ei[HQ];
  __shared__ float cand[128 * KNN];
  const int bs = blockIdx.x;
  const int b  = bs >> 10;
  const int i  = bs & (SQ - 1);
  const int t  = threadIdx.x;

  if (t < HQ) Ei[t] = E[(size_t)bs * HQ + t];
  __syncthreads();

  float best[KNN];
  #pragma unroll
  for (int q = 0; q < KNN; ++q) best[q] = BIGF;
  float worst = BIGF;
  const float sqi = sqv[bs];
  const int rowbase = b << 10;

  for (int j = t; j < i; j += 128) {
    const float4* ej  = reinterpret_cast<const float4*>(E + (size_t)(rowbase + j) * HQ);
    const float4* eiv = reinterpret_cast<const float4*>(Ei);
    float dot = 0.f;
    #pragma unroll
    for (int q = 0; q < HQ / 4; ++q) {
      float4 a = eiv[q]; float4 c = ej[q];
      dot += a.x * c.x + a.y * c.y + a.z * c.z + a.w * c.w;
    }
    float d2 = fmaxf(sqi + sqv[rowbase + j] - 2.f * dot, 1e-12f);
    if (d2 < worst) {
      bool rep = false;
      #pragma unroll
      for (int q = 0; q < KNN; ++q)
        if (!rep && best[q] == worst) { best[q] = d2; rep = true; }
      worst = best[0];
      #pragma unroll
      for (int q = 1; q < KNN; ++q) worst = fmaxf(worst, best[q]);
    }
  }
  #pragma unroll
  for (int q = 0; q < KNN; ++q) cand[t * KNN + q] = best[q];
  __syncthreads();

  for (int stride = 64; stride >= 1; stride >>= 1) {
    if (t < stride) {
      float out[KNN];
      #pragma unroll
      for (int k = 0; k < KNN; ++k) {
        float mn = BIGF; int mi = -1;
        for (int q = 0; q < 2 * KNN; ++q) {
          int idx = (q < KNN) ? (t * KNN + q) : ((t + stride) * KNN + (q - KNN));
          float v = cand[idx];
          if (v < mn) { mn = v; mi = idx; }
        }
        out[k] = mn;
        if (mi >= 0) cand[mi] = BIGF;
      }
      #pragma unroll
      for (int k = 0; k < KNN; ++k) cand[t * KNN + k] = out[k];
    }
    __syncthreads();
  }

  if (t == 0) {
    float s = 0.f;
    #pragma unroll
    for (int k = 0; k < KNN; ++k) {
      float d2 = cand[k];
      float d  = (d2 >= 0.5f * BIGF) ? BIGF : sqrtf(d2);
      dists[(size_t)bs * KNN + k] = d;
      s += d;
    }
    bm[bs] = s * (1.f / (float)KNN);
  }
}

// ---------------- sequential RunningMeanStd scan ----------------
__global__ void scan_kernel(const float* __restrict__ bm, float* __restrict__ means) {
  if (threadIdx.x != 0 || blockIdx.x != 0) return;
  float mean = 0.f, count = 1e-4f;
  for (int n = 0; n < BS; ++n) {
    float bmi = bm[n];
    bool  v   = (n & (SQ - 1)) >= KNN;
    float nc  = count + (float)KNN;
    float nm  = mean + (bmi - mean) * (float)KNN / nc;
    if (v) { mean = nm; count = nc; }
    means[n] = mean;
  }
}

// ---------------- null mask per batch ----------------
__global__ void null_kernel(const int* __restrict__ is_null,
                            int* __restrict__ firstArr, int* __restrict__ null_cnt) {
  __shared__ int sh_first[BQ];
  __shared__ int sh_has[BQ];
  int b = threadIdx.x;
  if (b < BQ) {
    int f = SQ, has = 0;
    for (int s = 0; s < SQ; ++s)
      if (is_null[b * SQ + s] != 0) { f = s; has = 1; break; }
    sh_first[b] = f;
    sh_has[b]   = has;
    firstArr[b] = has ? f : SQ;
  }
  __syncthreads();
  if (b == 0) {
    int cnt = 0;
    for (int q = 0; q < BQ; ++q)
      if (sh_has[q]) cnt += SQ - sh_first[q];
    *null_cnt = cnt;
  }
}

// ---------------- kernel-weighted pseudo-count reward ----------------
__global__ void reward_kernel(const float* __restrict__ dists,
                              const float* __restrict__ means,
                              const int* __restrict__ firstArr,
                              float* __restrict__ out) {
  int bs = blockIdx.x * blockDim.x + threadIdx.x;
  if (bs >= BS) return;
  int b = bs >> 10;
  int s = bs & (SQ - 1);
  float mean = means[bs];
  float ksum = 0.f;
  #pragma unroll
  for (int k = 0; k < KNN; ++k) {
    float sd = dists[(size_t)bs * KNN + k] / (mean + 1e-11f);
    sd = fmaxf(sd - 0.008f, 0.f);
    ksum += 1e-4f / (sd + 1e-4f);
  }
  float sv = sqrtf(fmaxf(ksum, 0.f)) + 0.001f;
  float r  = (sv > 8.f) ? 0.f : (1.f / sv);
  if (s < KNN) r = 0.f;
  if (s >= firstArr[b]) r = 0.f;
  out[bs] = r;
}

// ---------------- deterministic reduce -> mean_r ----------------
__global__ __launch_bounds__(256) void finalize_kernel(
    const float* __restrict__ rf, const int* __restrict__ null_cnt,
    float* __restrict__ out_scalar) {
  __shared__ float sh[256];
  int t = threadIdx.x;
  float s = 0.f;
  for (int i = t; i < BS; i += 256) s += rf[i];
  sh[t] = s;
  __syncthreads();
  for (int st = 128; st >= 1; st >>= 1) {
    if (t < st) sh[t] += sh[t + st];
    __syncthreads();
  }
  if (t == 0) out_scalar[0] = sh[0] / ((float)BS - (float)(*null_cnt));
}

// ---------------- host launcher ----------------
extern "C" void kernel_launch(void* const* d_in, const int* in_sizes, int n_in,
                              void* d_out, int out_size, void* d_ws, size_t ws_size,
                              hipStream_t stream) {
  (void)in_sizes; (void)n_in; (void)out_size; (void)ws_size;
  const float* obs     = (const float*)d_in[0];
  const int*   is_null = (const int*)d_in[1];
  const float* W0 = (const float*)d_in[2];
  const float* b0 = (const float*)d_in[3];
  const float* W1 = (const float*)d_in[4];
  const float* b1 = (const float*)d_in[5];
  const float* W2 = (const float*)d_in[6];
  const float* b2 = (const float*)d_in[7];

  char* ws = (char*)d_ws;
  size_t off = 0;
  auto alloc = [&](size_t bytes) -> void* {
    off = (off + 255) & ~(size_t)255;
    void* p = ws + off;
    off += bytes;
    return p;
  };

  unsigned short* bufA = (unsigned short*)alloc((size_t)BS * DQ * 2);  // obs bf16, reused as x2
  unsigned short* bufB = (unsigned short*)alloc((size_t)BS * 512 * 2); // x1 bf16
  unsigned short* w0t  = (unsigned short*)alloc((size_t)512 * 512 * 2); // [N,K]
  unsigned short* w1t  = (unsigned short*)alloc((size_t)512 * 512 * 2); // [N,K]
  unsigned short* w2t  = (unsigned short*)alloc((size_t)128 * 512 * 2); // [N,K]
  float* E     = (float*)alloc((size_t)BS * HQ * 4);
  float* sqv   = (float*)alloc((size_t)BS * 4);
  float* dists = (float*)alloc((size_t)BS * KNN * 4);
  float* bm    = (float*)alloc((size_t)BS * 4);
  float* means = (float*)alloc((size_t)BS * 4);
  int* firstArr = (int*)alloc(BQ * 4);
  int* null_cnt = (int*)alloc(4);

  // convert obs; transpose+convert weights to [N,K] bf16
  cvt_bf16_kernel<<<2048, 256, 0, stream>>>(obs, bufA, BS * DQ);
  transpose_bf16_kernel<<<dim3(512 / 32, 512 / 32), 256, 0, stream>>>(W0, w0t, 512, 512);
  transpose_bf16_kernel<<<dim3(512 / 32, 512 / 32), 256, 0, stream>>>(W1, w1t, 512, 512);
  transpose_bf16_kernel<<<dim3(128 / 32, 512 / 32), 256, 0, stream>>>(W2, w2t, 512, 128);

  // MLP: 3 WMMA GEMMs with fused bias+ReLU (TDM double-buffered tiles)
  dim3 g12(512 / NT, BS / MT);  // (4, 256)
  gemm_bias_relu<false><<<g12, 256, 0, stream>>>(bufA, w0t, b0, bufB, nullptr, BS, 512, 512);
  gemm_bias_relu<false><<<g12, 256, 0, stream>>>(bufB, w1t, b1, bufA, nullptr, BS, 512, 512);
  dim3 g3(128 / NT, BS / MT);   // (1, 256)
  gemm_bias_relu<true><<<g3, 256, 0, stream>>>(bufA, w2t, b2, nullptr, E, BS, 128, 512);

  // distances + top-K
  sq_kernel<<<BS / 256, 256, 0, stream>>>(E, sqv);
  dist_topk_kernel<<<BS, 128, 0, stream>>>(E, sqv, dists, bm);

  // sequential scan
  scan_kernel<<<1, 1, 0, stream>>>(bm, means);

  // null mask + reward + mean
  null_kernel<<<1, 32, 0, stream>>>(is_null, firstArr, null_cnt);
  reward_kernel<<<BS / 256, 256, 0, stream>>>(dists, means, firstArr, (float*)d_out);
  finalize_kernel<<<1, 256, 0, stream>>>((float*)d_out, null_cnt, (float*)d_out + BS);
}
